// MoE_23940147708255
// MI455X (gfx1250) — matrix operand order repaired and checked by reading, not compiled, not run on hardware
//
#include <hip/hip_runtime.h>
#include <math.h>

#define DIMX 512
#define HIDX 2048
#define NEXP 16
#define BB   8
#define NTOK 2048
#define CAPX 256
#define NTOT (BB*NTOK)          // 16384 tokens
#define THRESH 0.8f
#define LOSS_COEF 0.01f

typedef __attribute__((ext_vector_type(16))) __bf16 v16bf;
typedef __attribute__((ext_vector_type(8)))  __bf16 v8bf;
typedef __attribute__((ext_vector_type(8)))  float  v8f;

// ---------------- workspace layout (bytes) ----------------
constexpr size_t SZ_EIF32 = (size_t)NEXP*BB*CAPX*DIMX*4;   // 67 MB (reused as eo_f32)
constexpr size_t SZ_EIB   = (size_t)NEXP*BB*CAPX*DIMX*2;   // 33.5 MB
constexpr size_t SZ_W1T   = (size_t)NEXP*HIDX*DIMX*2;      // 33.5 MB  [E][H][D]
constexpr size_t SZ_W2T   = (size_t)NEXP*DIMX*HIDX*2;      // 33.5 MB  [E][D][H]
constexpr size_t SZ_HID   = (size_t)NEXP*BB*CAPX*HIDX*2;   // 134 MB
constexpr size_t SZ_WTS   = (size_t)NTOT*NEXP*4;           // 1 MB
constexpr size_t SZ_U32   = (size_t)NTOT*4;                // 64 KB each

constexpr size_t OFF_EIF32 = 0;
constexpr size_t OFF_EIB   = OFF_EIF32 + SZ_EIF32;
constexpr size_t OFF_W1T   = OFF_EIB   + SZ_EIB;
constexpr size_t OFF_W2T   = OFF_W1T   + SZ_W1T;
constexpr size_t OFF_HID   = OFF_W2T   + SZ_W2T;
constexpr size_t OFF_WTS   = OFF_HID   + SZ_HID;
constexpr size_t OFF_MASK0 = OFF_WTS   + SZ_WTS;
constexpr size_t OFF_MASKF = OFF_MASK0 + SZ_U32;
constexpr size_t OFF_POS   = OFF_MASKF + SZ_U32;
constexpr size_t OFF_SUMR  = OFF_POS   + SZ_U32;           // 128 floats
constexpr size_t OFF_SUMM  = OFF_SUMR  + 512;              // 128 floats

// ---------------- utility kernels ----------------
__global__ void k_zero_f32(float* __restrict__ p, size_t n) {
  size_t i = (size_t)blockIdx.x * blockDim.x + threadIdx.x;
  size_t s = (size_t)gridDim.x * blockDim.x;
  for (; i < n; i += s) p[i] = 0.f;
}

__global__ void k_f32_to_bf16(const float* __restrict__ in, __bf16* __restrict__ out, size_t n) {
  size_t i = (size_t)blockIdx.x * blockDim.x + threadIdx.x;
  size_t s = (size_t)gridDim.x * blockDim.x;
  for (; i < n; i += s) out[i] = (__bf16)in[i];
}

// in: [E][rows][cols] f32  ->  out: [E][cols][rows] bf16
__global__ void k_transpose_bf16(const float* __restrict__ in, __bf16* __restrict__ out,
                                 int rows, int cols) {
  size_t per = (size_t)rows * cols;
  size_t n = (size_t)NEXP * per;
  size_t i = (size_t)blockIdx.x * blockDim.x + threadIdx.x;
  size_t s = (size_t)gridDim.x * blockDim.x;
  for (; i < n; i += s) {
    size_t e = i / per, rem = i % per;
    size_t c = rem / rows, r = rem % rows;
    out[i] = (__bf16)in[(e * rows + r) * cols + c];
  }
}

// ---------------- gating: logits -> softmax -> threshold top-k ----------------
__global__ void k_gate(const float* __restrict__ x, const float* __restrict__ wg,
                       float* __restrict__ wts, unsigned* __restrict__ mask0,
                       float* __restrict__ sum_raw) {
  int wid = threadIdx.x >> 5, lane = threadIdx.x & 31;
  int t = blockIdx.x * 8 + wid;
  if (t >= NTOT) return;
  const float* xp = x + (size_t)t * DIMX;
  float acc[NEXP];
#pragma unroll
  for (int e = 0; e < NEXP; ++e) acc[e] = 0.f;
  for (int ko = 0; ko < DIMX / 32; ++ko) {
    float xv = xp[ko * 32 + lane];
    const float* wrow = wg + (size_t)(ko * 32 + lane) * NEXP;
#pragma unroll
    for (int e = 0; e < NEXP; ++e) acc[e] += xv * wrow[e];
  }
#pragma unroll
  for (int e = 0; e < NEXP; ++e)
#pragma unroll
    for (int off = 16; off >= 1; off >>= 1) acc[e] += __shfl_xor(acc[e], off, 32);

  if (lane == 0) {
    int b = t / NTOK;
    float mx = acc[0];
    for (int e = 1; e < NEXP; ++e) mx = fmaxf(mx, acc[e]);
    float p[NEXP]; float s = 0.f;
    for (int e = 0; e < NEXP; ++e) { p[e] = expf(acc[e] - mx); s += p[e]; }
    float inv = 1.f / s;
    for (int e = 0; e < NEXP; ++e) { p[e] *= inv; atomicAdd(&sum_raw[b * NEXP + e], p[e]); }
    // stable descending insertion sort (matches argsort(-raw))
    float sp[NEXP]; int ord[NEXP];
    for (int i = 0; i < NEXP; ++i) { sp[i] = p[i]; ord[i] = i; }
    for (int i = 1; i < NEXP; ++i) {
      float kv = sp[i]; int ko2 = ord[i]; int j = i - 1;
      while (j >= 0 && sp[j] < kv) { sp[j + 1] = sp[j]; ord[j + 1] = ord[j]; --j; }
      sp[j + 1] = kv; ord[j + 1] = ko2;
    }
    float cum = 0.f; int k = NEXP;
    for (int i = 0; i < NEXP; ++i) { cum += sp[i]; if (cum >= THRESH) { k = i + 1; break; } }
    float ssum = 0.f;
    for (int i = 0; i < k; ++i) ssum += sp[i];
    float rn = 1.f / ssum;
    float w[NEXP]; unsigned mb = 0;
    for (int e = 0; e < NEXP; ++e) w[e] = 0.f;
    for (int i = 0; i < k; ++i) { w[ord[i]] = sp[i] * rn; mb |= (1u << ord[i]); }
    for (int e = 0; e < NEXP; ++e) wts[(size_t)t * NEXP + e] = w[e];
    mask0[t] = mb;
  }
}

// ---------------- per-batch sequential capacity scan ----------------
__global__ void k_capacity(const unsigned* __restrict__ mask0, unsigned* __restrict__ maskf,
                           int* __restrict__ pos_tok, float* __restrict__ sum_mask) {
  int b = blockIdx.x;
  int lane = threadIdx.x;       // 32 lanes; lanes 0..15 own experts
  int e = lane;
  int cnt = 0, kept = 0;
  for (int n = 0; n < NTOK; ++n) {
    unsigned mbits = mask0[b * NTOK + n];
    int m = (e < NEXP) ? (int)((mbits >> e) & 1u) : 0;
    int keep = (m && cnt < CAPX) ? 1 : 0;      // pos = exclusive cumsum of ORIGINAL mask
    int pc = keep ? cnt : 0;                   // pos zeroed where capacity-dropped
    cnt += m;
    kept += keep;
    unsigned long long bal = __ballot(keep);
    int tot = pc;
#pragma unroll
    for (int off = 16; off >= 1; off >>= 1) tot += __shfl_xor(tot, off, 32);
    if (lane == 0) {
      maskf[b * NTOK + n] = (unsigned)(bal & 0xFFFFu);
      pos_tok[b * NTOK + n] = tot;             // summed across experts, as in reference
    }
  }
  if (e < NEXP) sum_mask[b * NEXP + e] = (float)kept;
}

// ---------------- dispatch scatter: ei[e,b,c,:] += x[b,n,:] ----------------
__global__ void k_dispatch(const float* __restrict__ x, const unsigned* __restrict__ maskf,
                           const int* __restrict__ pos_tok, float* __restrict__ ei) {
  int wid = threadIdx.x >> 5, lane = threadIdx.x & 31;
  int t = blockIdx.x * 8 + wid;
  if (t >= NTOT) return;
  unsigned mb = maskf[t];
  int c = pos_tok[t];
  if (mb == 0 || c >= CAPX) return;
  int b = t / NTOK;
  const float* xp = x + (size_t)t * DIMX;
  for (int ko = 0; ko < DIMX / 32; ++ko) {
    int d = ko * 32 + lane;
    float xv = xp[d];
    unsigned mm = mb;
    while (mm) {
      int e = __ffs(mm) - 1; mm &= mm - 1;
      atomicAdd(&ei[((((size_t)e * BB + b) * CAPX) + c) * DIMX + d], xv);
    }
  }
}

// ---------------- WMMA fragment loaders (contiguous k = 16h + s in both A and B) ----------------
__device__ inline v16bf load_frag_a(const __bf16* __restrict__ mat, int ld, int row0, int k0) {
  int lane = threadIdx.x & 31;
  int idx = lane & 15, half = lane >> 4;
  const __bf16* p = mat + (size_t)(row0 + idx) * ld + k0 + half * 16;
  v8bf lo = *(const v8bf*)(p);          // K = 16h .. 16h+7
  v8bf hi = *(const v8bf*)(p + 8);      // K = 16h+8 .. 16h+15
  v16bf f;
#pragma unroll
  for (int i = 0; i < 8; ++i) { f[i] = lo[i]; f[8 + i] = hi[i]; }
  return f;
}

__device__ inline v16bf load_frag_b(const __bf16* __restrict__ matT, int ld, int col0, int k0) {
  int lane = threadIdx.x & 31;
  int idx = lane & 15, half = lane >> 4;
  // B-matrix: lane-half holds 16 contiguous K values for its column
  return *(const v16bf*)(matT + (size_t)(col0 + idx) * ld + k0 + half * 16);
}

// ---------------- bf16 WMMA GEMM: C[M,N] = A[M,K] * Bt[N,K]^T ----------------
// 8 waves/block as 2(M) x 4(N); 64x32 tile per wave = 4x2 WMMA; block tile 128x128.
// DO_GELU=1: exact-erf GELU fused, bf16 output.  DO_GELU=0: f32 output.
template <int DO_GELU>
__global__ __launch_bounds__(256, 1)
void k_gemm(const __bf16* __restrict__ A, const __bf16* __restrict__ Bt,
            void* __restrict__ Cout, int M, int N, int K,
            size_t strideA, size_t strideB, size_t strideC) {
  int e = blockIdx.y;
  const __bf16* Ae = A + (size_t)e * strideA;
  const __bf16* Be = Bt + (size_t)e * strideB;
  int tilesN = N / 128;
  int bm = blockIdx.x / tilesN, bn = blockIdx.x % tilesN;
  int wid = threadIdx.x >> 5;
  int m0 = bm * 128 + (wid & 1) * 64;
  int n0 = bn * 128 + (wid >> 1) * 32;

  v8f acc[4][2];
#pragma unroll
  for (int i = 0; i < 4; ++i)
#pragma unroll
    for (int j = 0; j < 2; ++j)
#pragma unroll
      for (int r = 0; r < 8; ++r) acc[i][j][r] = 0.f;

  for (int k0 = 0; k0 < K; k0 += 32) {
    v16bf a[4], b[2];
#pragma unroll
    for (int i = 0; i < 4; ++i) a[i] = load_frag_a(Ae, K, m0 + 16 * i, k0);
#pragma unroll
    for (int j = 0; j < 2; ++j) b[j] = load_frag_b(Be, K, n0 + 16 * j, k0);
#pragma unroll
    for (int i = 0; i < 4; ++i)
#pragma unroll
      for (int j = 0; j < 2; ++j)
        acc[i][j] = __builtin_amdgcn_wmma_f32_16x16x32_bf16(false, a[i], false, b[j],
                                                            (short)0, acc[i][j], false, false);
  }

  int lane = threadIdx.x & 31;
  int idx = lane & 15, half = lane >> 4;
#pragma unroll
  for (int i = 0; i < 4; ++i)
#pragma unroll
    for (int j = 0; j < 2; ++j) {
      int mbase = m0 + i * 16 + half * 8;
      int col = n0 + j * 16 + idx;
#pragma unroll
      for (int r = 0; r < 8; ++r) {
        float v = acc[i][j][r];
        if (DO_GELU) {
          v = 0.5f * v * (1.f + erff(v * 0.70710678118654752f));
          ((__bf16*)Cout)[(size_t)e * strideC + (size_t)(mbase + r) * N + col] = (__bf16)v;
        } else {
          ((float*)Cout)[(size_t)e * strideC + (size_t)(mbase + r) * N + col] = v;
        }
      }
    }
}

// ---------------- combine gather: out[b,n,:] = sum_e w * eo[e,b,c,:] ----------------
__global__ void k_combine(const float* __restrict__ eo, const float* __restrict__ wts,
                          const unsigned* __restrict__ maskf, const int* __restrict__ pos_tok,
                          float* __restrict__ out) {
  int wid = threadIdx.x >> 5, lane = threadIdx.x & 31;
  int t = blockIdx.x * 8 + wid;
  if (t >= NTOT) return;
  unsigned mb = maskf[t];
  int c = pos_tok[t];
  int b = t / NTOK;
  bool act = (mb != 0u) && (c < CAPX);
  float4* outv = (float4*)(out + (size_t)t * DIMX);
#pragma unroll
  for (int ko = 0; ko < DIMX / 128; ++ko) {   // 4 iterations of float4 per lane
    int d4 = ko * 32 + lane;
    float4 v = make_float4(0.f, 0.f, 0.f, 0.f);
    if (act) {
      unsigned mm = mb;
      while (mm) {
        int e = __ffs(mm) - 1; mm &= mm - 1;
        float w = wts[(size_t)t * NEXP + e];
        const float4* ev = (const float4*)(eo + ((((size_t)e * BB + b) * CAPX) + c) * DIMX);
        float4 ex = ev[d4];
        v.x += w * ex.x; v.y += w * ex.y; v.z += w * ex.z; v.w += w * ex.w;
      }
    }
    outv[d4] = v;
  }
}

// ---------------- loss ----------------
__global__ void k_loss(const float* __restrict__ sum_raw, const float* __restrict__ sum_mask,
                       float* __restrict__ out_loss) {
  __shared__ float red[128];
  int t = threadIdx.x;
  float v = 0.f;
  if (t < BB * NEXP) v = (sum_raw[t] / (float)NTOK) * (sum_mask[t] / (float)NTOK);
  red[t] = v;
  __syncthreads();
  for (int s = 64; s > 0; s >>= 1) { if (t < s) red[t] += red[t + s]; __syncthreads(); }
  if (t == 0) *out_loss = (red[0] / (float)(BB * NEXP)) * (float)(NEXP * NEXP) * LOSS_COEF;
}

// ---------------- launcher ----------------
extern "C" void kernel_launch(void* const* d_in, const int* in_sizes, int n_in,
                              void* d_out, int out_size, void* d_ws, size_t ws_size,
                              hipStream_t stream) {
  const float* x  = (const float*)d_in[0];
  const float* wg = (const float*)d_in[1];
  const float* w1 = (const float*)d_in[2];
  const float* w2 = (const float*)d_in[3];
  char* ws = (char*)d_ws;
  float*    ei_f32 = (float*)(ws + OFF_EIF32);   // reused as eo_f32 after GEMM2
  __bf16*   ei_b   = (__bf16*)(ws + OFF_EIB);
  __bf16*   w1t    = (__bf16*)(ws + OFF_W1T);
  __bf16*   w2t    = (__bf16*)(ws + OFF_W2T);
  __bf16*   hid    = (__bf16*)(ws + OFF_HID);
  float*    wts    = (float*)(ws + OFF_WTS);
  unsigned* mask0  = (unsigned*)(ws + OFF_MASK0);
  unsigned* maskf  = (unsigned*)(ws + OFF_MASKF);
  int*      postk  = (int*)(ws + OFF_POS);
  float*    sum_r  = (float*)(ws + OFF_SUMR);
  float*    sum_m  = (float*)(ws + OFF_SUMM);
  float*    out    = (float*)d_out;

  // zero accumulators
  k_zero_f32<<<4096, 256, 0, stream>>>(ei_f32, SZ_EIF32 / 4);
  k_zero_f32<<<1, 256, 0, stream>>>(sum_r, 256);   // sum_r + sum_m contiguous

  // gating + capacity + dispatch
  k_gate<<<NTOT / 8, 256, 0, stream>>>(x, wg, wts, mask0, sum_r);
  k_capacity<<<BB, 32, 0, stream>>>(mask0, maskf, postk, sum_m);
  k_dispatch<<<NTOT / 8, 256, 0, stream>>>(x, maskf, postk, ei_f32);

  // precision conversion / weight transposes
  k_transpose_bf16<<<4096, 256, 0, stream>>>(w1, w1t, DIMX, HIDX); // [E][H][D]
  k_transpose_bf16<<<4096, 256, 0, stream>>>(w2, w2t, HIDX, DIMX); // [E][D][H]
  k_f32_to_bf16<<<4096, 256, 0, stream>>>(ei_f32, ei_b, (size_t)NEXP * BB * CAPX * DIMX);

  // GEMM1: [2048x512] x [512x2048] -> gelu -> bf16 hidden
  {
    dim3 grid((2048 / 128) * (2048 / 128), NEXP);   // 256 x 16
    k_gemm<1><<<grid, 256, 0, stream>>>(ei_b, w1t, hid, 2048, 2048, 512,
                                        (size_t)2048 * 512, (size_t)2048 * 512,
                                        (size_t)2048 * 2048);
  }
  // GEMM2: [2048x2048] x [2048x512] -> f32 eo (reuses ei_f32 region)
  {
    dim3 grid((2048 / 128) * (512 / 128), NEXP);    // 64 x 16
    k_gemm<0><<<grid, 256, 0, stream>>>(hid, w2t, ei_f32, 2048, 512, 2048,
                                        (size_t)2048 * 2048, (size_t)512 * 2048,
                                        (size_t)2048 * 512);
  }

  // combine + loss
  k_combine<<<NTOT / 8, 256, 0, stream>>>(ei_f32, wts, maskf, postk, out);
  k_loss<<<1, 128, 0, stream>>>(sum_r, sum_m, out + (out_size - 1));
  (void)in_sizes; (void)n_in; (void)ws_size;
}
